// MultiHeadedAttention_46978352283827
// MI455X (gfx1250) — compile-verified, matrix operand order checked by reading, not run on hardware
//
#include <hip/hip_runtime.h>

#define D_MODEL 1024
#define NHEAD   16
#define DK      64
#define BATCH   4
#define SEQ     2048
#define MROWS   (BATCH * SEQ)   // 8192

typedef __bf16 bf16;
typedef __attribute__((ext_vector_type(16))) __bf16 v16bf;
typedef __attribute__((ext_vector_type(8)))  float  v8f;

// ---------------------------------------------------------------------------
// CDNA5 async memory->LDS staging (ASYNCcnt path, CDNA5 ISA).
// global_load_async_to_lds_b128: vdst = 32-bit LDS byte offset, vaddr = 64-bit
// global address. Generic shared pointers on amdgcn carry the LDS offset in
// their low 32 bits.
// ---------------------------------------------------------------------------
__device__ __forceinline__ uint32_t lds_off_u32(const void* p) {
    return (uint32_t)(uintptr_t)p;
}

__device__ __forceinline__ void async_b128(uint32_t lds_byte_off, const void* gptr) {
    asm volatile("global_load_async_to_lds_b128 %0, %1, off"
                 :: "v"(lds_byte_off), "v"((uint64_t)(uintptr_t)gptr)
                 : "memory");
}

template <int N>
__device__ __forceinline__ void wait_async() {
    asm volatile("s_wait_asynccnt %0" :: "n"(N) : "memory");
}

// ---------------------------------------------------------------------------
// Load a 16x32 bf16 WMMA A/B fragment from an LDS tile laid out row-major
// with leading dimension `ld` (multiple of 8 -> b128-aligned chunks).
// Per ISA 7.12.2: lane l (0-15) = matrix row rbase+l; lane half selects the
// K sub-chunks [hi*8..hi*8+7] and [16+hi*8..16+hi*8+7].
// ---------------------------------------------------------------------------
__device__ __forceinline__ v16bf load_frag16(const bf16* tile, int ld, int rbase, int koff) {
    const int lane = threadIdx.x & 31;
    const int l    = lane & 15;
    const int hi   = lane >> 4;
    const bf16* p  = tile + (rbase + l) * ld + koff + hi * 8;
    union { uint4 u[2]; v16bf f; } t;
    t.u[0] = *reinterpret_cast<const uint4*>(p);
    t.u[1] = *reinterpret_cast<const uint4*>(p + 16);
    return t.f;
}

__device__ __forceinline__ v8f wmma_bf16(v16bf a, v16bf b, v8f c) {
    return __builtin_amdgcn_wmma_f32_16x16x32_bf16(false, a, false, b, (short)0, c, false, false);
}

// ---------------------------------------------------------------------------
// fp32 -> bf16 conversion, vectorized 4 elements/thread
// ---------------------------------------------------------------------------
__global__ void cvt_kernel(const float* __restrict__ in, bf16* __restrict__ out, int n4) {
    int i      = blockIdx.x * blockDim.x + threadIdx.x;
    int stride = gridDim.x * blockDim.x;
    for (; i < n4; i += stride) {
        float4 f = reinterpret_cast<const float4*>(in)[i];
        union { bf16 h[4]; uint64_t u; } o;
        o.h[0] = (bf16)f.x; o.h[1] = (bf16)f.y; o.h[2] = (bf16)f.z; o.h[3] = (bf16)f.w;
        reinterpret_cast<uint64_t*>(out)[i] = o.u;
    }
}

// ---------------------------------------------------------------------------
// C[m,n] = sum_k A[m,k] * W[n,k] + bias[n]  (A: 8192x1024 bf16, W: 1024x1024)
// mode 0: write bf16 head-split [B][H][S][DK]     (QKV projections)
// mode 1: write fp32 (acc + bias + resid[m,n])    (output projection pre-LN)
// 256 threads (8 waves), 128x128 block tile, K-step 32, double-buffered
// async-to-LDS staging with a peeled, branch-free steady-state loop.
// Wave w owns rows (w&3)*32..+31, cols (w>>2)*64..+63.
// ---------------------------------------------------------------------------
__global__ void __launch_bounds__(256)
gemm_bf16(const bf16* __restrict__ A, const bf16* __restrict__ W,
          const float* __restrict__ bias, const float* __restrict__ resid,
          bf16* __restrict__ out_b, float* __restrict__ out_f, int mode) {
    __shared__ __align__(16) bf16 As[2][128][40];
    __shared__ __align__(16) bf16 Bs[2][128][40];

    const int tid   = threadIdx.x;
    const int wave  = tid >> 5;
    const int lane  = tid & 31;
    const int l     = lane & 15;
    const int hi    = lane >> 4;
    const int wrow  = (wave & 3) * 32;
    const int wcol  = (wave >> 2) * 64;
    const int mtile = blockIdx.y * 128;
    const int ntile = blockIdx.x * 128;

    const int row = tid >> 1;         // 0..127
    const int seg = (tid & 1) * 16;   // 0 or 16 elements

    uint32_t laA[2] = { lds_off_u32(&As[0][row][seg]), lds_off_u32(&As[1][row][seg]) };
    uint32_t laB[2] = { lds_off_u32(&Bs[0][row][seg]), lds_off_u32(&Bs[1][row][seg]) };
    const bf16* pA = A + (size_t)(mtile + row) * D_MODEL + seg;
    const bf16* pB = W + (size_t)(ntile + row) * D_MODEL + seg;

    auto issue = [&](int buf) {
        async_b128(laA[buf],      pA);
        async_b128(laA[buf] + 16, pA + 8);
        async_b128(laB[buf],      pB);
        async_b128(laB[buf] + 16, pB + 8);
        pA += 32; pB += 32;
    };

    v8f acc[2][4] = {};
    auto compute = [&](int buf) {
        v16bf a0 = load_frag16(&As[buf][0][0], 40, wrow, 0);
        v16bf a1 = load_frag16(&As[buf][0][0], 40, wrow + 16, 0);
#pragma unroll
        for (int j = 0; j < 4; ++j) {
            v16bf bf = load_frag16(&Bs[buf][0][0], 40, wcol + j * 16, 0);
            acc[0][j] = wmma_bf16(a0, bf, acc[0][j]);
            acc[1][j] = wmma_bf16(a1, bf, acc[1][j]);
        }
    };

    constexpr int NSTEPS = D_MODEL / 32;   // 32
    issue(0);
    int cur = 0;
    for (int i = 0; i < NSTEPS - 1; ++i) {       // branch-free steady state
        issue(cur ^ 1);        // prefetch next slab (freed by last iter's barrier)
        wait_async<4>();       // current slab's 4 in-order asyncs complete
        __syncthreads();
        compute(cur);
        __syncthreads();
        cur ^= 1;
    }
    wait_async<0>();
    __syncthreads();
    compute(cur);

#pragma unroll
    for (int mi = 0; mi < 2; ++mi) {
#pragma unroll
        for (int j = 0; j < 4; ++j) {
            const int   n  = ntile + wcol + j * 16 + l;
            const float bn = bias[n];
#pragma unroll
            for (int r = 0; r < 8; ++r) {
                const int m = mtile + wrow + mi * 16 + hi * 8 + r;
                float v = acc[mi][j][r] + bn;
                if (mode == 0) {
                    const int bb = m >> 11, s = m & (SEQ - 1);
                    const int hh = n >> 6,  d = n & (DK - 1);
                    out_b[(((size_t)bb * NHEAD + hh) * SEQ + s) * DK + d] = (bf16)v;
                } else {
                    v += resid[(size_t)m * D_MODEL + n];
                    out_f[(size_t)m * D_MODEL + n] = v;
                }
            }
        }
    }
}

// ---------------------------------------------------------------------------
// Flash attention: block = one (b,h), 64 query rows; 4 waves x 16 rows each.
// K tiles double-buffered via async-to-LDS; V tiles double-buffered via
// b128 register loads + transposed LDS scatter. One barrier per key-tile.
// ---------------------------------------------------------------------------
__global__ void __launch_bounds__(128)
attn_kernel(const bf16* __restrict__ Qh, const bf16* __restrict__ Kh,
            const bf16* __restrict__ Vh, bf16* __restrict__ ctx) {
    __shared__ __align__(16) bf16 Qs [64][72];
    __shared__ __align__(16) bf16 Ks [2][64][72];
    __shared__ __align__(16) bf16 Vts[2][64][72];   // transposed: [d][key]
    __shared__ __align__(16) bf16 Ps [4][16][64];   // per-wave P tile

    const int tid  = threadIdx.x;
    const int wave = tid >> 5;
    const int lane = tid & 31;
    const int l    = lane & 15;
    const int hi   = lane >> 4;

    const int bh = blockIdx.x;            // b*NHEAD + h
    const int b  = bh >> 4;
    const int h  = bh & (NHEAD - 1);
    const int q0 = blockIdx.y * 64;

    const bf16* Qp = Qh + (size_t)bh * SEQ * DK;

    const int row = tid >> 1;             // 0..63
    const int seg = (tid & 1) * 32;       // 0 or 32 elements (64 bytes)

    uint32_t lsK[2] = { lds_off_u32(&Ks[0][row][seg]), lds_off_u32(&Ks[1][row][seg]) };
    const bf16* pK = Kh + (size_t)bh * SEQ * DK + (size_t)row * DK + seg;
    const bf16* pV = Vh + (size_t)bh * SEQ * DK + (size_t)row * DK + seg;

    auto issueK = [&](int buf) {
#pragma unroll
        for (int i = 0; i < 4; ++i) async_b128(lsK[buf] + i * 16, pK + i * 8);
        pK += 64 * DK;
    };
    auto loadV = [&](uint4* vr) {
        const uint4* g = reinterpret_cast<const uint4*>(pV);
#pragma unroll
        for (int i = 0; i < 4; ++i) vr[i] = g[i];
        pV += 64 * DK;
    };
    auto storeV = [&](int buf, const uint4* vr) {   // transpose scatter
        union { uint4 u[4]; bf16 h[32]; } t;
#pragma unroll
        for (int i = 0; i < 4; ++i) t.u[i] = vr[i];
#pragma unroll
        for (int i = 0; i < 32; ++i) Vts[buf][seg + i][row] = t.h[i];
    };

    {   // stage Q tile (row-major) via async copies
        const bf16* g  = Qp + (size_t)(q0 + row) * DK + seg;
        uint32_t    ls = lds_off_u32(&Qs[row][seg]);
#pragma unroll
        for (int i = 0; i < 4; ++i) async_b128(ls + i * 16, g + i * 8);
    }

    v8f   o[4] = {};
    float rmax[8], rsum[8];
#pragma unroll
    for (int r = 0; r < 8; ++r) { rmax[r] = -3.0e38f; rsum[r] = 0.f; }

    // prologue: first K tile async + first V tile through registers
    issueK(0);
    {
        uint4 vr[4];
        loadV(vr);
        storeV(0, vr);
    }
    wait_async<0>();          // Q + K0 in LDS
    __syncthreads();          // V0 ds-stores fenced too

    const v16bf aq0 = load_frag16(&Qs[0][0], 72, wave * 16, 0);    // d = 0..31
    const v16bf aq1 = load_frag16(&Qs[0][0], 72, wave * 16, 32);   // d = 32..63

    auto attn_step = [&](int buf) {
        // scores = Q K^T / sqrt(dk)
        v8f sc[4];
#pragma unroll
        for (int j = 0; j < 4; ++j) {
            v16bf bk0 = load_frag16(&Ks[buf][0][0], 72, j * 16, 0);
            v16bf bk1 = load_frag16(&Ks[buf][0][0], 72, j * 16, 32);
            v8f c = {};
            c = wmma_bf16(aq0, bk0, c);
            c = wmma_bf16(aq1, bk1, c);
            sc[j] = c * 0.125f;   // 1/sqrt(64)
        }

        // online softmax (half-wave owns each row; xor masks 1..8 stay in half)
#pragma unroll
        for (int r = 0; r < 8; ++r) {
            float cmax = fmaxf(fmaxf(sc[0][r], sc[1][r]), fmaxf(sc[2][r], sc[3][r]));
#pragma unroll
            for (int m = 8; m >= 1; m >>= 1) cmax = fmaxf(cmax, __shfl_xor(cmax, m, 32));
            const float nmax = fmaxf(rmax[r], cmax);
            const float corr = __expf(rmax[r] - nmax);
            rmax[r] = nmax;
            float ls = 0.f;
#pragma unroll
            for (int j = 0; j < 4; ++j) {
                const float p = __expf(sc[j][r] - nmax);
                ls += p;
                Ps[wave][hi * 8 + r][j * 16 + l] = (bf16)p;
            }
#pragma unroll
            for (int m = 8; m >= 1; m >>= 1) ls += __shfl_xor(ls, m, 32);
            rsum[r] = rsum[r] * corr + ls;
#pragma unroll
            for (int j = 0; j < 4; ++j) o[j][r] *= corr;
        }

        // O += P V  (wave-internal LDS store->load is kept in order)
        const v16bf ap0 = load_frag16(&Ps[wave][0][0], 64, 0, 0);    // keys 0..31
        const v16bf ap1 = load_frag16(&Ps[wave][0][0], 64, 0, 32);   // keys 32..63
#pragma unroll
        for (int j = 0; j < 4; ++j) {
            v16bf bv0 = load_frag16(&Vts[buf][0][0], 72, j * 16, 0);
            v16bf bv1 = load_frag16(&Vts[buf][0][0], 72, j * 16, 32);
            o[j] = wmma_bf16(ap0, bv0, o[j]);
            o[j] = wmma_bf16(ap1, bv1, o[j]);
        }
    };

    int cur = 0;
    for (int kc = 0; kc < SEQ / 64 - 1; ++kc) {
        issueK(cur ^ 1);            // next K tile: buffer freed by prev barrier
        uint4 vr[4];
        loadV(vr);                  // next V tile into registers (overlaps WMMA)
        attn_step(cur);
        storeV(cur ^ 1, vr);        // transpose-scatter next V tile
        wait_async<0>();
        __syncthreads();            // single barrier per key tile
        cur ^= 1;
    }
    attn_step(cur);

    // write context re-merged as [B][S][H*DK] (bf16) for the output GEMM
#pragma unroll
    for (int r = 0; r < 8; ++r) {
        const int   s   = q0 + wave * 16 + hi * 8 + r;
        const float inv = 1.f / rsum[r];
#pragma unroll
        for (int j = 0; j < 4; ++j) {
            const int d = j * 16 + l;
            ctx[((size_t)b * SEQ + s) * D_MODEL + h * DK + d] = (bf16)(o[j][r] * inv);
        }
    }
}

// ---------------------------------------------------------------------------
// Row LayerNorm: out = (x - mu) * rsqrt(var + eps) * g + b
// ---------------------------------------------------------------------------
__global__ void __launch_bounds__(256)
ln_kernel(const float* __restrict__ x, const float* __restrict__ g,
          const float* __restrict__ beta, float* __restrict__ out) {
    const int row = blockIdx.x;
    const int tid = threadIdx.x;
    const float* xr = x + (size_t)row * D_MODEL;

    float v[4];
    float s = 0.f;
#pragma unroll
    for (int i = 0; i < 4; ++i) { v[i] = xr[tid + i * 256]; s += v[i]; }

    __shared__ float red[256];
    red[tid] = s; __syncthreads();
    for (int st = 128; st > 0; st >>= 1) { if (tid < st) red[tid] += red[tid + st]; __syncthreads(); }
    const float mu = red[0] * (1.f / D_MODEL);
    __syncthreads();

    float ss = 0.f;
#pragma unroll
    for (int i = 0; i < 4; ++i) { const float d = v[i] - mu; ss += d * d; }
    red[tid] = ss; __syncthreads();
    for (int st = 128; st > 0; st >>= 1) { if (tid < st) red[tid] += red[tid + st]; __syncthreads(); }
    const float rstd = rsqrtf(red[0] * (1.f / D_MODEL) + 1e-5f);

#pragma unroll
    for (int i = 0; i < 4; ++i) {
        const int c = tid + i * 256;
        out[(size_t)row * D_MODEL + c] = (v[i] - mu) * rstd * g[c] + beta[c];
    }
}

// ---------------------------------------------------------------------------
extern "C" void kernel_launch(void* const* d_in, const int* in_sizes, int n_in,
                              void* d_out, int out_size, void* d_ws, size_t ws_size,
                              hipStream_t stream) {
    (void)in_sizes; (void)n_in; (void)out_size; (void)ws_size;

    const float* q   = (const float*)d_in[0];
    const float* k   = (const float*)d_in[1];
    const float* v   = (const float*)d_in[2];
    const float* Wq  = (const float*)d_in[3];
    const float* bq  = (const float*)d_in[4];
    const float* Wk  = (const float*)d_in[5];
    const float* bk  = (const float*)d_in[6];
    const float* Wv  = (const float*)d_in[7];
    const float* bv  = (const float*)d_in[8];
    const float* Wo  = (const float*)d_in[9];
    const float* bo  = (const float*)d_in[10];
    const float* lng = (const float*)d_in[11];
    const float* lnb = (const float*)d_in[12];
    float* out = (float*)d_out;

    size_t off = 0;
    auto alloc = [&](size_t bytes) {
        void* p = (char*)d_ws + off;
        off += (bytes + 255) & ~(size_t)255;
        return p;
    };
    const size_t NE = (size_t)MROWS * D_MODEL;            // 8M elements
    const size_t WE = (size_t)D_MODEL * D_MODEL;          // 1M elements

    bf16* qb  = (bf16*)alloc(NE * 2);
    bf16* kb  = (bf16*)alloc(NE * 2);
    bf16* vb  = (bf16*)alloc(NE * 2);
    bf16* wqb = (bf16*)alloc(WE * 2);
    bf16* wkb = (bf16*)alloc(WE * 2);
    bf16* wvb = (bf16*)alloc(WE * 2);
    bf16* wob = (bf16*)alloc(WE * 2);
    bf16* Qh  = (bf16*)alloc(NE * 2);                     // [B][H][S][DK]
    bf16* Kh  = (bf16*)alloc(NE * 2);
    bf16* Vh  = (bf16*)alloc(NE * 2);
    bf16* ctx = (bf16*)alloc(NE * 2);                     // [B][S][D_MODEL]
    float* opre = (float*)alloc(NE * 4);                  // pre-LayerNorm

    // 1) fp32 -> bf16
    cvt_kernel<<<1024, 256, 0, stream>>>(q,  qb,  (int)(NE / 4));
    cvt_kernel<<<1024, 256, 0, stream>>>(k,  kb,  (int)(NE / 4));
    cvt_kernel<<<1024, 256, 0, stream>>>(v,  vb,  (int)(NE / 4));
    cvt_kernel<<<256,  256, 0, stream>>>(Wq, wqb, (int)(WE / 4));
    cvt_kernel<<<256,  256, 0, stream>>>(Wk, wkb, (int)(WE / 4));
    cvt_kernel<<<256,  256, 0, stream>>>(Wv, wvb, (int)(WE / 4));
    cvt_kernel<<<256,  256, 0, stream>>>(Wo, wob, (int)(WE / 4));

    // 2) QKV projections (head-split bf16 output)
    dim3 ggrid(D_MODEL / 128, MROWS / 128);   // (8, 64)
    gemm_bf16<<<ggrid, 256, 0, stream>>>(qb, wqb, bq, nullptr, Qh, nullptr, 0);
    gemm_bf16<<<ggrid, 256, 0, stream>>>(kb, wkb, bk, nullptr, Kh, nullptr, 0);
    gemm_bf16<<<ggrid, 256, 0, stream>>>(vb, wvb, bv, nullptr, Vh, nullptr, 0);

    // 3) flash attention
    attn_kernel<<<dim3(BATCH * NHEAD, SEQ / 64), 128, 0, stream>>>(Qh, Kh, Vh, ctx);

    // 4) output projection + bias + residual (fp32)
    gemm_bf16<<<ggrid, 256, 0, stream>>>(ctx, wob, bo, q, nullptr, opre, 1);

    // 5) LayerNorm -> d_out
    ln_kernel<<<MROWS, 256, 0, stream>>>(opre, lng, lnb, out);
}
